// RNN_45131516346882
// MI455X (gfx1250) — compile-verified, matrix operand order checked by reading, not run on hardware
//
#include <hip/hip_runtime.h>
#include <hip/hip_bf16.h>

// ---------------------------------------------------------------------------
// Masked LSTM for MI455X (gfx1250).
//   Phase 1: x_proj = input @ W_x + b      (big bf16 WMMA GEMM, bf16 scratch)
//   Phase 2: persistent recurrence kernel, W_h chunk resident in LDS,
//            v_wmma_f32_16x16x32_bf16 per step, device-wide barrier per step.
// Both GEMM operands are staged in LDS so each lane's fragment data is
// K-contiguous -> every fragment load is exactly 2x ds_load_b128.
// ---------------------------------------------------------------------------

typedef __attribute__((ext_vector_type(16))) __bf16 v16bf;
typedef __attribute__((ext_vector_type(8)))  float  v8f;

constexpr int Tn = 2048;
constexpr int Bn = 32;
constexpr int Dn = 512;
constexpr int Hn = 512;
constexpr int Gn = 4 * Hn;                 // 2048 gate columns
constexpr int Mrows = Tn * Bn;             // 65536

constexpr int NBLK_REC = 16;               // persistent workgroups
constexpr int CHUNK_H  = Hn / NBLK_REC;    // 32 h-columns per block

// workspace layout (bytes)
constexpr size_t XPROJ_ELEMS = (size_t)Mrows * Gn;        // bf16
constexpr size_t XPROJ_BYTES = XPROJ_ELEMS * 2;           // 256 MB
constexpr size_t MBUF_ELEMS  = (size_t)Bn * Hn;           // one m buffer (bf16)
constexpr size_t MBUF_BYTES  = MBUF_ELEMS * 2 * 2;        // double buffered

// recurrence kernel dynamic LDS (all bf16 row strides keep 16B alignment)
constexpr int WH_LD   = 520;   // whT transposed: [128 cols][512 K + pad]
constexpr int MT_LD   = 520;   // mT: [32 rows][512 K + pad]
constexpr int GT_LD   = 132;   // gates: [32][128 + pad] f32
constexpr size_t REC_LDS_BYTES =
    (size_t)(128 * WH_LD + Bn * MT_LD) * 2 +               // whT + mT (bf16)
    (size_t)(Bn * GT_LD + Bn * CHUNK_H) * 4;               // gates + c-state

// ---------------------------------------------------------------------------
// bf16 <-> f32 helpers (bit-level, round-to-nearest-even on pack)
// ---------------------------------------------------------------------------
__device__ __forceinline__ __bf16 f2bf(float f) {
  union { float f; unsigned u; } x; x.f = f;
  unsigned r = x.u + 0x7FFFu + ((x.u >> 16) & 1u);
  union { unsigned short s; __bf16 b; } y; y.s = (unsigned short)(r >> 16);
  return y.b;
}
__device__ __forceinline__ float bf2f(__bf16 h) {
  union { unsigned short s; __bf16 b; } y; y.b = h;
  union { unsigned u; float f; } x; x.u = ((unsigned)y.s) << 16;
  return x.f;
}

// ---------------------------------------------------------------------------
// Fragment loader (wave32; layout per CDNA5 ISA §7.12.2).
// Works for A (row-major [row][K]) and for B staged TRANSPOSED ([col][K]):
// lane l -> major index (l&15); VGPR v, half h -> K =
//   (v>>2)*16 + (l>>4)*8 + (v&3)*2 + h
// i.e. two 16-byte contiguous chunks per lane -> 2x ds_load_b128.
// ---------------------------------------------------------------------------
union Frag16 {
  struct { uint4 lo, hi; } q;
  v16bf v;
};

__device__ __forceinline__ v16bf load_frag(const __bf16* base, int ld,
                                           int major0, int k0) {
  const int l  = threadIdx.x & 31;
  const int r  = major0 + (l & 15);
  const int hi = l >> 4;
  const __bf16* p = base + r * ld + k0 + hi * 8;
  Frag16 f;
  f.q.lo = *(const uint4*)(p);        // K = k0+hi*8   .. +7
  f.q.hi = *(const uint4*)(p + 16);   // K = k0+16+hi*8.. +7
  return f.v;
}

__device__ __forceinline__ v8f wmma_bf16(v16bf a, v16bf b, v8f c) {
  return __builtin_amdgcn_wmma_f32_16x16x32_bf16(
      /*neg_a=*/false, a, /*neg_b=*/false, b,
      /*c_mod=*/(short)0, c, /*reuse_a=*/false, /*reuse_b=*/false);
}

// ---------------------------------------------------------------------------
// init: zero m double-buffer and barrier state (re-run every launch so the
// persistent kernel is deterministic across graph replays)
// ---------------------------------------------------------------------------
__global__ void rnn_init(unsigned short* mbuf, int* sync) {
  int tid = blockIdx.x * blockDim.x + threadIdx.x;
  int stride = gridDim.x * blockDim.x;
  for (int i = tid; i < (int)(2 * MBUF_ELEMS); i += stride) mbuf[i] = 0;
  if (tid < 2) sync[tid] = 0;
}

// ---------------------------------------------------------------------------
// Phase 1: x_proj[M=65536, N=2048] = input[M,512] @ W_x[512,2048] + b
// 128x128 tile per block, 256 threads (8 waves), bf16 WMMA, f32 accum.
// ---------------------------------------------------------------------------
__global__ __launch_bounds__(256) void rnn_xproj(
    const float* __restrict__ inp, const float* __restrict__ Wx,
    const float* __restrict__ bias, __bf16* __restrict__ xproj) {
  __shared__ __bf16 sA[128][40];     // A tile: 128 rows x 32 K (+pad)
  __shared__ __bf16 sBt[128][40];    // B tile TRANSPOSED: 128 cols x 32 K

  const int tid   = threadIdx.x;
  const int row0  = blockIdx.y * 128;
  const int col0  = blockIdx.x * 128;
  const int wid   = tid >> 5;
  const int waveM = wid & 3;        // 4 groups of 32 rows
  const int waveN = wid >> 2;       // 2 groups of 64 cols

  v8f acc[2][4] = {};

  for (int k0 = 0; k0 < Dn; k0 += 32) {
    for (int i = tid; i < 128 * 32; i += 256) {
      int r = i >> 5, c = i & 31;
      sA[r][c] = f2bf(inp[(size_t)(row0 + r) * Dn + k0 + c]);
    }
    for (int i = tid; i < 32 * 128; i += 256) {
      int k = i >> 7, c = i & 127;   // coalesced global read, LDS scatter
      sBt[c][k] = f2bf(Wx[(size_t)(k0 + k) * Gn + col0 + c]);
    }
    __syncthreads();

    v16bf afr[2], bfr[4];
#pragma unroll
    for (int mf = 0; mf < 2; ++mf)
      afr[mf] = load_frag(&sA[0][0], 40, waveM * 32 + mf * 16, 0);
#pragma unroll
    for (int nf = 0; nf < 4; ++nf)
      bfr[nf] = load_frag(&sBt[0][0], 40, waveN * 64 + nf * 16, 0);
#pragma unroll
    for (int mf = 0; mf < 2; ++mf)
#pragma unroll
      for (int nf = 0; nf < 4; ++nf)
        acc[mf][nf] = wmma_bf16(afr[mf], bfr[nf], acc[mf][nf]);
    __syncthreads();
  }

  // epilogue: += bias, pack bf16, store (C/D layout: lanes 0-15 M=v, N=lane;
  // lanes 16-31 M=8+v, N=lane-16)
  const int l    = tid & 31;
  const int n    = l & 15;
  const int moff = (l >> 4) * 8;
#pragma unroll
  for (int mf = 0; mf < 2; ++mf)
#pragma unroll
    for (int nf = 0; nf < 4; ++nf)
#pragma unroll
      for (int v = 0; v < 8; ++v) {
        int row = row0 + waveM * 32 + mf * 16 + moff + v;
        int col = col0 + waveN * 64 + nf * 16 + n;
        xproj[(size_t)row * Gn + col] = f2bf(acc[mf][nf][v] + bias[col]);
      }
}

// ---------------------------------------------------------------------------
// device-wide barrier (sense/generation) for the 16 persistent blocks
// ---------------------------------------------------------------------------
__device__ __forceinline__ void grid_sync(int* counter, int* gen, int nblocks) {
  __syncthreads();
  if (threadIdx.x == 0) {
    __threadfence();
    int g = __hip_atomic_load(gen, __ATOMIC_ACQUIRE, __HIP_MEMORY_SCOPE_AGENT);
    if (__hip_atomic_fetch_add(counter, 1, __ATOMIC_ACQ_REL,
                               __HIP_MEMORY_SCOPE_AGENT) == nblocks - 1) {
      __hip_atomic_store(counter, 0, __ATOMIC_RELAXED, __HIP_MEMORY_SCOPE_AGENT);
      __hip_atomic_fetch_add(gen, 1, __ATOMIC_RELEASE, __HIP_MEMORY_SCOPE_AGENT);
    } else {
      while (__hip_atomic_load(gen, __ATOMIC_ACQUIRE,
                               __HIP_MEMORY_SCOPE_AGENT) == g) {
        __builtin_amdgcn_s_sleep(1);
      }
    }
  }
  __syncthreads();
}

// ---------------------------------------------------------------------------
// Phase 2: persistent masked-LSTM recurrence.
// Block j owns h in [j*32, j*32+32): gate columns {h, H+h, 2H+h, 3H+h}.
// W_h chunk (512x128 bf16, stored transposed) lives in LDS for all 2048
// steps; c-state in LDS. m is double-buffered (bf16) in global; one grid
// barrier per timestep.
// ---------------------------------------------------------------------------
__global__ __launch_bounds__(256) void rnn_rec(
    const __bf16* __restrict__ xproj, const float* __restrict__ pad,
    const float* __restrict__ Wh, float* __restrict__ out,
    __bf16* __restrict__ mbuf, int* __restrict__ sync) {
  extern __shared__ char smem[];
  __bf16* whT = (__bf16*)smem;                 // [128 cols][WH_LD] transposed
  __bf16* mT  = whT + 128 * WH_LD;             // [32][MT_LD]  full m (bf16)
  float*  gT  = (float*)(mT + Bn * MT_LD);     // [32][GT_LD]  gate pre-acts
  float*  cSt = gT + Bn * GT_LD;               // [32*32]      c-state chunk

  const int tid = threadIdx.x;
  const int j   = blockIdx.x;
  const int h0  = j * CHUNK_H;

  // one-time: stage this block's W_h columns, transposed ([col][K]).
  // local col c = g*32+cc  ->  global gate col g*Hn + h0 + cc
  for (int i = tid; i < 512 * 128; i += 256) {
    int k = i >> 7, c = i & 127;
    int g = c >> 5, cc = c & 31;
    whT[c * WH_LD + k] = f2bf(Wh[(size_t)k * Gn + g * Hn + h0 + cc]);
  }
  for (int i = tid; i < Bn * CHUNK_H; i += 256) cSt[i] = 0.f;

  const int wid   = tid >> 5;        // 8 waves, wave w -> gate cols [16w,16w+16)
  const int l     = tid & 31;
  const int nlane = l & 15;
  const int moff  = (l >> 4) * 8;

  for (int t = 0; t < Tn; ++t) {
    const __bf16* mSrc = mbuf + (size_t)(t & 1) * MBUF_ELEMS;
    __bf16*       mDst = mbuf + (size_t)((t + 1) & 1) * MBUF_ELEMS;

    __syncthreads();
    // stage m [32 x 512] bf16 -> LDS as uint4 (2048 16B chunks, 8/thread)
    for (int i = tid; i < (Bn * Hn) / 8; i += 256) {
      int r = i >> 6, c = i & 63;    // 64 uint4 per row
      *(uint4*)(mT + r * MT_LD + c * 8) =
          *(const uint4*)(mSrc + (size_t)r * Hn + c * 8);
    }
    __syncthreads();

    // gates_pre[:, 16w:16w+16] = m @ Wh_chunk[:, 16w:16w+16]
    v8f acc0 = {}, acc1 = {};
#pragma unroll
    for (int k0 = 0; k0 < Hn; k0 += 32) {
      v16bf bfr = load_frag(whT, WH_LD, wid * 16, k0);
      v16bf a0  = load_frag(mT, MT_LD, 0,  k0);
      v16bf a1  = load_frag(mT, MT_LD, 16, k0);
      acc0 = wmma_bf16(a0, bfr, acc0);
      acc1 = wmma_bf16(a1, bfr, acc1);
    }
#pragma unroll
    for (int v = 0; v < 8; ++v) {
      gT[(moff + v) * GT_LD + wid * 16 + nlane]      = acc0[v];
      gT[(16 + moff + v) * GT_LD + wid * 16 + nlane] = acc1[v];
    }
    __syncthreads();

    // elementwise LSTM update for this block's 32x32 (b,h) chunk
    for (int i = tid; i < Bn * CHUNK_H; i += 256) {
      int b = i >> 5, hh = i & 31;
      int h = h0 + hh;
      size_t xb = ((size_t)t * Bn + b) * Gn;
      float gi = bf2f(xproj[xb + 0 * Hn + h]) + gT[b * GT_LD + 0 + hh];
      float gf = bf2f(xproj[xb + 1 * Hn + h]) + gT[b * GT_LD + 32 + hh];
      float gg = bf2f(xproj[xb + 2 * Hn + h]) + gT[b * GT_LD + 64 + hh];
      float go = bf2f(xproj[xb + 3 * Hn + h]) + gT[b * GT_LD + 96 + hh];

      float si = 1.f / (1.f + __expf(-gi));
      float sf = 1.f / (1.f + __expf(-gf));
      float so = 1.f / (1.f + __expf(-go));
      float tg = tanhf(gg);

      float c_old = cSt[i];
      float m_old = bf2f(mT[b * MT_LD + h]);
      float c_new = sf * c_old + si * tg;
      float m_new = so * tanhf(c_new);

      float p  = pad[t * Bn + b];
      float m2 = m_new + (m_old - m_new) * p;
      float c2 = c_new + (c_old - c_new) * p;

      cSt[i] = c2;
      out[((size_t)t * Bn + b) * Hn + h] = m2;
      mDst[b * Hn + h] = f2bf(m2);

      if (t + 1 < Tn)   // pull next step's xp slice toward the caches
        __builtin_prefetch(&xproj[((size_t)(t + 1) * Bn + b) * Gn + h], 0, 1);

      if (t == Tn - 1) {
        out[(size_t)Tn * Bn * Hn + (size_t)b * Hn + h] = m2;
        out[(size_t)Tn * Bn * Hn + (size_t)Bn * Hn + (size_t)b * Hn + h] = c2;
      }
    }
    grid_sync(sync, sync + 1, (int)gridDim.x);
  }
}

// ---------------------------------------------------------------------------
extern "C" void kernel_launch(void* const* d_in, const int* in_sizes, int n_in,
                              void* d_out, int out_size, void* d_ws,
                              size_t ws_size, hipStream_t stream) {
  const float* inp  = (const float*)d_in[0];   // [T,B,D]
  const float* pad  = (const float*)d_in[1];   // [T,B,1]
  const float* Wx   = (const float*)d_in[2];   // [D,4H]
  const float* Wh   = (const float*)d_in[3];   // [H,4H]
  const float* bias = (const float*)d_in[4];   // [4H]
  float* out = (float*)d_out;                  // [T,B,H] + [B,H] + [B,H]

  char* ws = (char*)d_ws;
  __bf16* xproj = (__bf16*)ws;                              // bf16 [M,4H]
  __bf16* mbuf  = (__bf16*)(ws + XPROJ_BYTES);              // 2x bf16 [B,H]
  int*    sync  = (int*)(ws + XPROJ_BYTES + MBUF_BYTES);    // counter + gen

  (void)in_sizes; (void)n_in; (void)out_size; (void)ws_size;

  // allow >64KB dynamic LDS for the persistent kernel (320KB/WGP on CDNA5)
  hipFuncSetAttribute((const void*)rnn_rec,
                      hipFuncAttributeMaxDynamicSharedMemorySize,
                      (int)REC_LDS_BYTES);

  rnn_init<<<64, 256, 0, stream>>>((unsigned short*)mbuf, sync);

  dim3 gX(Gn / 128, Mrows / 128);   // 16 x 512 tiles
  rnn_xproj<<<gX, 256, 0, stream>>>(inp, Wx, bias, xproj);

  rnn_rec<<<NBLK_REC, 256, REC_LDS_BYTES, stream>>>(xproj, pad, Wh, out, mbuf,
                                                    sync);
}
